// SemanticsModulatedAttention_57028575756687
// MI455X (gfx1250) — matrix operand order repaired and checked by reading, not compiled, not run on hardware
//
#include <hip/hip_runtime.h>
#include <cstdint>
#include <cstddef>

// ---------------- constants ----------------
constexpr int cB   = 8;
constexpr int cT   = 1024;
constexpr int cD   = 1024;
constexpr int cH   = 8;
constexpr int cDH  = 128;
constexpr int cNT  = 77;      // text tokens
constexpr int cLT  = 768;     // text feature dim
constexpr int cKR  = 1024;    // K*Rt retrieval tokens
constexpr int cNTOK= 2125;    // 77 + 1024 + 1024
constexpr int cE   = 2048;
constexpr int cD2  = 2048;
constexpr int cKPAD= 2144;    // 2125 padded to multiple of 32
constexpr float NEGV = -1000000.0f;

typedef __attribute__((ext_vector_type(16))) __bf16 v16bf;
typedef __attribute__((ext_vector_type(8)))  float  v8f;

union FragU { uint4 u4[2]; v16bf v; };

__device__ inline unsigned short f2bf(float f) {
  unsigned u = __builtin_bit_cast(unsigned, f);
  unsigned r = u + 0x7FFFu + ((u >> 16) & 1u);
  return (unsigned short)(r >> 16);
}

__device__ inline void blk_reduce(float* sh, float& v) {
  int tid = threadIdx.x;
  sh[tid] = v; __syncthreads();
  for (int s = 128; s > 0; s >>= 1) {
    if (tid < s) sh[tid] += sh[tid + s];
    __syncthreads();
  }
  v = sh[0]; __syncthreads();
}

// ============================================================================
// FAST WMMA GEMM (full tiles: M,N multiples of 128; K multiple of 32)
// Double-buffered LDS fed by GLOBAL_LOAD_ASYNC_TO_LDS_B128 (ASYNCcnt).
// Block: 256 threads = 8 waves (4 along M x 2 along N), block tile 128x128.
// ============================================================================
__global__ __launch_bounds__(256)
void wmma_gemm_async_kernel(const unsigned short* __restrict__ A,
                            const unsigned short* __restrict__ Bt,
                            float* __restrict__ C,
                            const float* __restrict__ bias,
                            const float* __restrict__ resid,
                            int K, int lda, int ldb, int ldc,
                            long long strideA, long long strideB, long long strideC)
{
  const int z = blockIdx.z;
  A  += (size_t)z * strideA;
  Bt += (size_t)z * strideB;
  C  += (size_t)z * strideC;

  const int tid = threadIdx.x;
  const int bm0 = blockIdx.y * 128;
  const int bn0 = blockIdx.x * 128;

  __shared__ unsigned short lsA[2][128][40];   // 80B row stride; 16B-aligned chunks
  __shared__ unsigned short lsB[2][128][40];

  const int wave = tid >> 5;
  const int lane = tid & 31;
  const int wm = (wave >> 1) * 32;
  const int wn = (wave & 1) * 64;
  const int lm = lane & 15;
  const int hi = lane >> 4;

  const int ldrow  = tid >> 1;         // 0..127
  const int ldhalf = (tid & 1) * 16;   // 0 or 16 ushorts

  const unsigned short* gA = A  + (size_t)(bm0 + ldrow) * lda + ldhalf;
  const unsigned short* gB = Bt + (size_t)(bn0 + ldrow) * ldb + ldhalf;

  // LDS byte addresses (low 32 bits of shared-aperture generic address == LDS offset)
  unsigned aL[2], bL[2];
#pragma unroll
  for (int s = 0; s < 2; ++s) {
    aL[s] = (unsigned)(uintptr_t)&lsA[s][ldrow][ldhalf];
    bL[s] = (unsigned)(uintptr_t)&lsB[s][ldrow][ldhalf];
  }

  v8f acc[2][4];
#pragma unroll
  for (int i = 0; i < 2; ++i)
#pragma unroll
    for (int j = 0; j < 4; ++j)
      acc[i][j] = (v8f){0.f,0.f,0.f,0.f,0.f,0.f,0.f,0.f};

  const int nk = K >> 5;

  // prologue: stage tile 0 into buffer 0 (2 async-to-LDS per wave)
  {
    const unsigned short* pa = gA;
    const unsigned short* pb = gB;
    asm volatile("global_load_async_to_lds_b128 %0, %1, off" :: "v"(aL[0]), "v"(pa) : "memory");
    asm volatile("global_load_async_to_lds_b128 %0, %1, off" :: "v"(bL[0]), "v"(pb) : "memory");
  }

  int buf = 0;
  for (int kt = 0; kt < nk; ++kt) {
    if (kt + 1 < nk) {
      // stream next tile into the other buffer while this one computes
      const unsigned short* pa = gA + (size_t)(kt + 1) * 32;
      const unsigned short* pb = gB + (size_t)(kt + 1) * 32;
      asm volatile("global_load_async_to_lds_b128 %0, %1, off" :: "v"(aL[buf ^ 1]), "v"(pa) : "memory");
      asm volatile("global_load_async_to_lds_b128 %0, %1, off" :: "v"(bL[buf ^ 1]), "v"(pb) : "memory");
      asm volatile("s_wait_asynccnt 0x2" ::: "memory");   // current tile landed; next still in flight
    } else {
      asm volatile("s_wait_asynccnt 0x0" ::: "memory");
    }
    __syncthreads();

    FragU a[2], b[4];
    const int ak0 = hi ? 16 : 0;    // A lanes<16: K 0..15; lanes>=16: K 16..31
#pragma unroll
    for (int i = 0; i < 2; ++i) {
      const unsigned short* p = &lsA[buf][wm + i * 16 + lm][0];
      a[i].u4[0] = *(const uint4*)(p + ak0);
      a[i].u4[1] = *(const uint4*)(p + ak0 + 8);
    }
    const int bk0 = hi ? 16 : 0;    // B lanes<16: K 0..15; lanes>=16: K 16..31
#pragma unroll
    for (int j = 0; j < 4; ++j) {
      const unsigned short* p = &lsB[buf][wn + j * 16 + lm][0];
      b[j].u4[0] = *(const uint4*)(p + bk0);
      b[j].u4[1] = *(const uint4*)(p + bk0 + 8);
    }
#pragma unroll
    for (int i = 0; i < 2; ++i)
#pragma unroll
      for (int j = 0; j < 4; ++j)
        acc[i][j] = __builtin_amdgcn_wmma_f32_16x16x32_bf16(
            false, a[i].v, false, b[j].v, (short)0, acc[i][j], false, false);

    __syncthreads();   // all waves done reading buf before it is refilled
    buf ^= 1;
  }

  // Epilogue (no bounds checks: full tiles). C layout: VGPR r -> M = r + 8*hi, N = lane&15.
#pragma unroll
  for (int i = 0; i < 2; ++i) {
    const int rowBase = bm0 + wm + i * 16 + hi * 8;
#pragma unroll
    for (int j = 0; j < 4; ++j) {
      const int col = bn0 + wn + j * 16 + lm;
      const float badd = bias ? bias[col] : 0.f;
#pragma unroll
      for (int r = 0; r < 8; ++r) {
        const int row = rowBase + r;
        float v = acc[i][j][r] + badd;
        if (resid) v += resid[(size_t)row * ldc + col];
        C[(size_t)row * ldc + col] = v;
      }
    }
  }
}

// ============================================================================
// GUARDED WMMA GEMM (handles M not multiple of 128; used for the text GEMMs)
// ============================================================================
__global__ __launch_bounds__(256)
void wmma_gemm_kernel(const unsigned short* __restrict__ A,
                      const unsigned short* __restrict__ Bt,
                      float* __restrict__ C,
                      const float* __restrict__ bias,
                      const float* __restrict__ resid,
                      int M, int N, int K, int lda, int ldb, int ldc,
                      long long strideA, long long strideB, long long strideC)
{
  const int z = blockIdx.z;
  A  += (size_t)z * strideA;
  Bt += (size_t)z * strideB;
  C  += (size_t)z * strideC;

  const int tid = threadIdx.x;
  const int bm0 = blockIdx.y * 128;
  const int bn0 = blockIdx.x * 128;

  __shared__ unsigned short lsA[128][40];
  __shared__ unsigned short lsB[128][40];

  const int wave = tid >> 5;
  const int lane = tid & 31;
  const int wm = (wave >> 1) * 32;
  const int wn = (wave & 1) * 64;
  const int lm = lane & 15;
  const int hi = lane >> 4;

  v8f acc[2][4];
#pragma unroll
  for (int i = 0; i < 2; ++i)
#pragma unroll
    for (int j = 0; j < 4; ++j)
      acc[i][j] = (v8f){0.f,0.f,0.f,0.f,0.f,0.f,0.f,0.f};

  const int ldrow  = tid >> 1;
  const int ldhalf = (tid & 1) * 16;
  const int aRow = bm0 + ldrow;
  const int bRow = bn0 + ldrow;

  for (int k0 = 0; k0 < K; k0 += 32) {
    uint4 va = {0u,0u,0u,0u};
    uint4 vb = {0u,0u,0u,0u};
    if (aRow < M) va = *(const uint4*)(A  + (size_t)aRow * lda + k0 + ldhalf);
    if (bRow < N) vb = *(const uint4*)(Bt + (size_t)bRow * ldb + k0 + ldhalf);
    if (k0 + 32 < K) {   // look-ahead prefetch (global_prefetch_b8)
      if (aRow < M) __builtin_prefetch(A  + (size_t)aRow * lda + k0 + 32 + ldhalf, 0, 1);
      if (bRow < N) __builtin_prefetch(Bt + (size_t)bRow * ldb + k0 + 32 + ldhalf, 0, 1);
    }
    __syncthreads();
    *(uint4*)&lsA[ldrow][ldhalf] = va;
    *(uint4*)&lsB[ldrow][ldhalf] = vb;
    __syncthreads();

    FragU a[2], b[4];
    const int ak0 = hi ? 16 : 0;
#pragma unroll
    for (int i = 0; i < 2; ++i) {
      const unsigned short* p = &lsA[wm + i * 16 + lm][0];
      a[i].u4[0] = *(const uint4*)(p + ak0);
      a[i].u4[1] = *(const uint4*)(p + ak0 + 8);
    }
    const int bk0 = hi ? 16 : 0;
#pragma unroll
    for (int j = 0; j < 4; ++j) {
      const unsigned short* p = &lsB[wn + j * 16 + lm][0];
      b[j].u4[0] = *(const uint4*)(p + bk0);
      b[j].u4[1] = *(const uint4*)(p + bk0 + 8);
    }
#pragma unroll
    for (int i = 0; i < 2; ++i)
#pragma unroll
      for (int j = 0; j < 4; ++j)
        acc[i][j] = __builtin_amdgcn_wmma_f32_16x16x32_bf16(
            false, a[i].v, false, b[j].v, (short)0, acc[i][j], false, false);
  }

#pragma unroll
  for (int i = 0; i < 2; ++i) {
    const int rowBase = bm0 + wm + i * 16 + hi * 8;
#pragma unroll
    for (int j = 0; j < 4; ++j) {
      const int col = bn0 + wn + j * 16 + lm;
      if (col >= N) continue;
      const float badd = bias ? bias[col] : 0.f;
#pragma unroll
      for (int r = 0; r < 8; ++r) {
        const int row = rowBase + r;
        if (row < M) {
          float v = acc[i][j][r] + badd;
          if (resid) v += resid[(size_t)row * ldc + col];
          C[(size_t)row * ldc + col] = v;
        }
      }
    }
  }
}

// ---------------- LayerNorm -> bf16 ----------------
__global__ __launch_bounds__(256)
void ln_rows_kernel(const float* __restrict__ in, const float* __restrict__ g,
                    const float* __restrict__ bb, unsigned short* __restrict__ out, int W)
{
  __shared__ float sh[256];
  const int row = blockIdx.x;
  const float* r = in + (size_t)row * W;
  float vals[8];
  float s = 0.f, s2 = 0.f;
  int it = 0;
  for (int c = threadIdx.x; c < W; c += 256) { float v = r[c]; vals[it++] = v; s += v; s2 += v * v; }
  blk_reduce(sh, s);
  blk_reduce(sh, s2);
  const float mean = s / W;
  const float rstd = rsqrtf(s2 / W - mean * mean + 1e-5f);
  unsigned short* o = out + (size_t)row * W;
  it = 0;
  for (int c = threadIdx.x; c < W; c += 256)
    o[c] = f2bf((vals[it++] - mean) * rstd * g[c] + bb[c]);
}

// LN of concat(re_motion, re_text broadcast) rows (W = 2048)
__global__ __launch_bounds__(256)
void ln_refeat_kernel(const float* __restrict__ re_motion, const float* __restrict__ re_text,
                      const float* __restrict__ g, const float* __restrict__ bb,
                      unsigned short* __restrict__ out)
{
  __shared__ float sh[256];
  const int row = blockIdx.x;            // b*1024 + j
  const int b = row >> 10, j = row & 1023;
  const int k2 = j >> 9, rr = j & 511;
  const float* mot = re_motion + (((size_t)(b * 2 + k2)) * 512 + rr) * 1024;
  const float* txt = re_text + ((size_t)(b * 2 + k2)) * 1024;
  float vals[8];
  float s = 0.f, s2 = 0.f;
  int it = 0;
  for (int c = threadIdx.x; c < cD2; c += 256) {
    float v = (c < 1024) ? mot[c] : txt[c - 1024];
    vals[it++] = v; s += v; s2 += v * v;
  }
  blk_reduce(sh, s);
  blk_reduce(sh, s2);
  const float mean = s / cD2;
  const float rstd = rsqrtf(s2 / cD2 - mean * mean + 1e-5f);
  unsigned short* o = out + (size_t)row * cD2;
  it = 0;
  for (int c = threadIdx.x; c < cD2; c += 256)
    o[c] = f2bf((vals[it++] - mean) * rstd * g[c] + bb[c]);
}

// LN(y)*(1+scale)+shift -> silu -> bf16
__global__ __launch_bounds__(256)
void ln_film_silu_kernel(const float* __restrict__ y, const float* __restrict__ g,
                         const float* __restrict__ bb, const float* __restrict__ eo,
                         unsigned short* __restrict__ out)
{
  __shared__ float sh[256];
  const int row = blockIdx.x;            // b*T + t
  const int b = row / cT;
  const float* r = y + (size_t)row * cD;
  float vals[4];
  float s = 0.f, s2 = 0.f;
  int it = 0;
  for (int c = threadIdx.x; c < cD; c += 256) { float v = r[c]; vals[it++] = v; s += v; s2 += v * v; }
  blk_reduce(sh, s);
  blk_reduce(sh, s2);
  const float mean = s / cD;
  const float rstd = rsqrtf(s2 / cD - mean * mean + 1e-5f);
  unsigned short* o = out + (size_t)row * cD;
  it = 0;
  for (int c = threadIdx.x; c < cD; c += 256) {
    float nv = (vals[it++] - mean) * rstd * g[c] + bb[c];
    float f = nv * (1.f + eo[(size_t)b * cD2 + c]) + eo[(size_t)b * cD2 + cD + c];
    o[c] = f2bf(f / (1.f + __expf(-f)));
  }
}

// ---------------- weight convert + transpose: out[n*K+k] = bf16(in[k*N+n]) ----------------
__global__ void convtrans_kernel(const float* __restrict__ in, unsigned short* __restrict__ out,
                                 int K, int N)
{
  size_t idx = (size_t)blockIdx.x * 256 + threadIdx.x;
  if (idx >= (size_t)K * N) return;
  int n = (int)(idx / K), k = (int)(idx % K);
  out[idx] = f2bf(in[(size_t)k * N + n]);
}

// ---------------- masks ----------------
__global__ void keymask_kernel(float* __restrict__ key, const int* __restrict__ cond,
                               const float* __restrict__ re_mask, const float* __restrict__ src_mask)
{
  size_t idx = (size_t)blockIdx.x * 256 + threadIdx.x;
  if (idx >= (size_t)cB * cNTOK * cD) return;
  int b = (int)(idx / ((size_t)cNTOK * cD));
  int n = (int)((idx / cD) % cNTOK);
  int ct = cond[b];
  float add;
  if (n < cNT) {
    add = (1.f - (float)((ct % 10) > 0)) * NEGV;
  } else if (n < cNT + cKR) {
    float rm = re_mask[(size_t)b * cKR + (n - cNT)];
    add = (1.f - (float)((ct / 10) > 0)) * NEGV + (1.f - rm) * NEGV;
  } else {
    add = (1.f - src_mask[(size_t)b * cT + (n - cNT - cKR)]) * NEGV;
  }
  key[idx] += add;
}

__global__ void valmask_kernel(float* __restrict__ val, const int* __restrict__ cond,
                               const float* __restrict__ re_mask, const float* __restrict__ src_mask)
{
  size_t idx = (size_t)blockIdx.x * 256 + threadIdx.x;
  if (idx >= (size_t)cB * cNTOK * cD) return;
  int b = (int)(idx / ((size_t)cNTOK * cD));
  int n = (int)((idx / cD) % cNTOK);
  int ct = cond[b];
  float m;
  if (n < cNT) {
    m = (float)((ct % 10) > 0);
  } else if (n < cNT + cKR) {
    m = (float)((ct / 10) > 0) * re_mask[(size_t)b * cKR + (n - cNT)];
  } else {
    m = src_mask[(size_t)b * cT + (n - cNT - cKR)];
  }
  val[idx] *= m;
}

// ---------------- softmax over head dim (q) -> q_bf[b][h][t][d] ----------------
__global__ __launch_bounds__(256)
void qsoftmax_kernel(const float* __restrict__ query, unsigned short* __restrict__ qbf)
{
  const int gw = blockIdx.x * 8 + (threadIdx.x >> 5);   // row id over B*T*H
  const int lane = threadIdx.x & 31;
  const int h = gw % cH;
  const int t = (gw / cH) % cT;
  const int b = gw / (cH * cT);
  const float* r = query + ((size_t)(b * cT + t)) * cD + h * cDH;
  float v[4];
  float mx = -3.0e38f;
#pragma unroll
  for (int i = 0; i < 4; ++i) { v[i] = r[lane + 32 * i]; mx = fmaxf(mx, v[i]); }
#pragma unroll
  for (int o = 16; o > 0; o >>= 1) mx = fmaxf(mx, __shfl_xor(mx, o, 32));
  float s = 0.f;
#pragma unroll
  for (int i = 0; i < 4; ++i) { v[i] = __expf(v[i] - mx); s += v[i]; }
#pragma unroll
  for (int o = 16; o > 0; o >>= 1) s += __shfl_xor(s, o, 32);
  const float inv = 1.f / s;
  unsigned short* o = qbf + (((size_t)(b * cH + h)) * cT + t) * cDH;
#pragma unroll
  for (int i = 0; i < 4; ++i) o[lane + 32 * i] = f2bf(v[i] * inv);
}

// ---------------- softmax over tokens (k) -> kt_bf[b][h][d][n] ----------------
__global__ __launch_bounds__(256)
void ksoftmax_kernel(const float* __restrict__ key, unsigned short* __restrict__ kt)
{
  const int col = blockIdx.x * 256 + threadIdx.x;   // channel 0..1023
  const int b = blockIdx.y;
  const float* base = key + (size_t)b * cNTOK * cD + col;
  float mx = -3.0e38f;
  for (int n = 0; n < cNTOK; ++n) mx = fmaxf(mx, base[(size_t)n * cD]);
  float s = 0.f;
  for (int n = 0; n < cNTOK; ++n) s += __expf(base[(size_t)n * cD] - mx);
  const float inv = 1.f / s;
  const int h = col >> 7, dd = col & 127;
  unsigned short* o = kt + (((size_t)(b * cH + h)) * cDH + dd) * cKPAD;
  for (int n = 0; n < cNTOK; ++n) o[n] = f2bf(__expf(base[(size_t)n * cD] - mx) * inv);
}

// ---------------- v transpose -> vt_bf[b][h][l][n] ----------------
__global__ void vtranspose_kernel(const float* __restrict__ value, unsigned short* __restrict__ vt)
{
  size_t idx = (size_t)blockIdx.x * 256 + threadIdx.x;
  if (idx >= (size_t)cB * cNTOK * cD) return;
  int col = (int)(idx % cD);
  size_t bn = idx / cD;
  int n = (int)(bn % cNTOK);
  int b = (int)(bn / cNTOK);
  int h = col >> 7, dd = col & 127;
  vt[(((size_t)(b * cH + h)) * cDH + dd) * cKPAD + n] = f2bf(value[idx]);
}

// ---------------- att transpose: attT_bf[z][l][d] = bf16(att[z][d][l]) ----------------
__global__ void atttrans_kernel(const float* __restrict__ att, unsigned short* __restrict__ attT)
{
  size_t idx = (size_t)blockIdx.x * 256 + threadIdx.x;
  if (idx >= (size_t)cB * cH * cDH * cDH) return;
  int z = (int)(idx / (cDH * cDH));
  int rem = (int)(idx % (cDH * cDH));
  int l = rem / cDH, d = rem % cDH;
  attT[idx] = f2bf(att[(size_t)z * cDH * cDH + (size_t)d * cDH + l]);
}

// ---------------- y permute: y[b][t][h*128+l] = ytmp[(b*H+h)][t][l] ----------------
__global__ void ypermute_kernel(const float* __restrict__ ytmp, float* __restrict__ y)
{
  size_t idx = (size_t)blockIdx.x * 256 + threadIdx.x;
  if (idx >= (size_t)cB * cH * cT * cDH) return;
  int l = (int)(idx % cDH);
  int t = (int)((idx / cDH) % cT);
  int h = (int)((idx / ((size_t)cDH * cT)) % cH);
  int b = (int)(idx / ((size_t)cDH * cT * cH));
  y[((size_t)(b * cT + t)) * cD + h * cDH + l] = ytmp[idx];
}

// ---------------- silu(emb) ----------------
__global__ void siluemb_kernel(const float* __restrict__ emb, float* __restrict__ se)
{
  size_t idx = (size_t)blockIdx.x * 256 + threadIdx.x;
  if (idx >= (size_t)cB * cE) return;
  float v = emb[idx];
  se[idx] = v / (1.f + __expf(-v));
}

// ---------------- eo = silu(emb) @ We + be   (tiny M=8 matvec) ----------------
__global__ void eo_kernel(const float* __restrict__ se, const float* __restrict__ We,
                          const float* __restrict__ be, float* __restrict__ eo)
{
  const int j = blockIdx.x * 256 + threadIdx.x;   // 0..2047
  const int b = blockIdx.y;
  const float* s = se + (size_t)b * cE;
  float acc = be[j];
  for (int e = 0; e < cE; ++e) acc += s[e] * We[(size_t)e * cD2 + j];
  eo[(size_t)b * cD2 + j] = acc;
}

// ---------------- host ----------------
static inline size_t cdiv(size_t a, size_t b) { return (a + b - 1) / b; }

extern "C" void kernel_launch(void* const* d_in, const int* in_sizes, int n_in,
                              void* d_out, int out_size, void* d_ws, size_t ws_size,
                              hipStream_t stream)
{
  const float* x        = (const float*)d_in[0];
  const float* xf       = (const float*)d_in[1];
  const float* emb      = (const float*)d_in[2];
  const float* src_mask = (const float*)d_in[3];
  const int*   cond     = (const int*)d_in[4];
  const float* re_motion= (const float*)d_in[5];
  const float* re_text  = (const float*)d_in[6];
  const float* re_mask  = (const float*)d_in[7];
  const float* ln_g = (const float*)d_in[8],  *ln_b = (const float*)d_in[9];
  const float* tln_g= (const float*)d_in[10], *tln_b= (const float*)d_in[11];
  const float* rn1_g= (const float*)d_in[12], *rn1_b= (const float*)d_in[13];
  const float* rn2_g= (const float*)d_in[14], *rn2_b= (const float*)d_in[15];
  const float* Wq = (const float*)d_in[16], *bq = (const float*)d_in[17];
  const float* Wkt= (const float*)d_in[18], *bkt= (const float*)d_in[19];
  const float* Wvt= (const float*)d_in[20], *bvt= (const float*)d_in[21];
  const float* Wkm= (const float*)d_in[22], *bkm= (const float*)d_in[23];
  const float* Wvm= (const float*)d_in[24], *bvm= (const float*)d_in[25];
  const float* Wkr= (const float*)d_in[26], *bkr= (const float*)d_in[27];
  const float* Wvr= (const float*)d_in[28], *bvr= (const float*)d_in[29];
  const float* We = (const float*)d_in[30], *be = (const float*)d_in[31];
  const float* sln_g=(const float*)d_in[32], *sln_b=(const float*)d_in[33];
  const float* Wo = (const float*)d_in[34], *bo = (const float*)d_in[35];
  float* out = (float*)d_out;

  char* wp = (char*)d_ws;
  size_t off = 0;
  auto alloc = [&](size_t bytes) -> void* {
    void* p = wp + off;
    off = (off + bytes + 255) & ~(size_t)255;
    return p;
  };

  unsigned short* norm_x_bf   = (unsigned short*)alloc((size_t)cB*cT*cD*2);
  unsigned short* norm_xf_bf  = (unsigned short*)alloc((size_t)cB*cNT*cLT*2);
  unsigned short* norm_rf_bf  = (unsigned short*)alloc((size_t)cB*cKR*cD2*2);
  unsigned short* norm_rm_bf  = (unsigned short*)alloc((size_t)cB*cKR*cD*2);
  unsigned short* WqT  = (unsigned short*)alloc((size_t)cD*cD*2);
  unsigned short* WktT = (unsigned short*)alloc((size_t)cD*cLT*2);
  unsigned short* WvtT = (unsigned short*)alloc((size_t)cD*cLT*2);
  unsigned short* WkmT = (unsigned short*)alloc((size_t)cD*cD*2);
  unsigned short* WvmT = (unsigned short*)alloc((size_t)cD*cD*2);
  unsigned short* WkrT = (unsigned short*)alloc((size_t)cD*cD2*2);
  unsigned short* WvrT = (unsigned short*)alloc((size_t)cD*cD*2);
  unsigned short* WoT  = (unsigned short*)alloc((size_t)cD*cD*2);
  float* query_f = (float*)alloc((size_t)cB*cT*cD*4);
  float* key_f   = (float*)alloc((size_t)cB*cNTOK*cD*4);
  float* value_f = (float*)alloc((size_t)cB*cNTOK*cD*4);
  unsigned short* q_bf  = (unsigned short*)alloc((size_t)cB*cH*cT*cDH*2);
  unsigned short* kt_bf = (unsigned short*)alloc((size_t)cB*cH*cDH*cKPAD*2);
  unsigned short* vt_bf = (unsigned short*)alloc((size_t)cB*cH*cDH*cKPAD*2);
  float* att_f  = (float*)alloc((size_t)cB*cH*cDH*cDH*4);
  unsigned short* attT_bf = (unsigned short*)alloc((size_t)cB*cH*cDH*cDH*2);
  float* eo_f = (float*)alloc((size_t)cB*cD2*4);
  float* se_f = (float*)alloc((size_t)cB*cE*4);
  // aliases (consumed-before-reuse)
  float* ytmp_f = query_f;                 // query consumed by qsoftmax before y GEMM
  float* y_f    = key_f;                   // key consumed by ksoftmax before y permute
  unsigned short* h_bf = norm_x_bf;        // norm_x consumed by projection GEMMs before film

  const int TPB = 256;
  // fast path: M,N multiples of 128, K multiple of 32
  auto gemm = [&](const unsigned short* A, const unsigned short* Bt, float* C,
                  const float* bias, const float* resid,
                  int M, int N, int K, int lda, int ldb, int ldc,
                  long long sA, long long sB, long long sC, int batch) {
    dim3 g((unsigned)cdiv(N, 128), (unsigned)cdiv(M, 128), (unsigned)batch);
    if ((M % 128 == 0) && (N % 128 == 0)) {
      hipLaunchKernelGGL(wmma_gemm_async_kernel, g, dim3(TPB), 0, stream,
                         A, Bt, C, bias, resid, K, lda, ldb, ldc, sA, sB, sC);
    } else {
      hipLaunchKernelGGL(wmma_gemm_kernel, g, dim3(TPB), 0, stream,
                         A, Bt, C, bias, resid, M, N, K, lda, ldb, ldc, sA, sB, sC);
    }
  };

  // 1) weights -> bf16 transposed
  auto ctrans = [&](const float* W, unsigned short* Wt, int K, int N) {
    size_t tot = (size_t)K * N;
    hipLaunchKernelGGL(convtrans_kernel, dim3((unsigned)cdiv(tot, TPB)), dim3(TPB), 0, stream, W, Wt, K, N);
  };
  ctrans(Wq,  WqT,  cD,  cD);
  ctrans(Wkt, WktT, cLT, cD);
  ctrans(Wvt, WvtT, cLT, cD);
  ctrans(Wkm, WkmT, cD,  cD);
  ctrans(Wvm, WvmT, cD,  cD);
  ctrans(Wkr, WkrT, cD2, cD);
  ctrans(Wvr, WvrT, cD,  cD);
  ctrans(Wo,  WoT,  cD,  cD);

  // 2) layernorms -> bf16 operands
  hipLaunchKernelGGL(ln_rows_kernel, dim3(cB*cT), dim3(TPB), 0, stream, x,  ln_g,  ln_b,  norm_x_bf,  cD);
  hipLaunchKernelGGL(ln_rows_kernel, dim3(cB*cNT), dim3(TPB), 0, stream, xf, tln_g, tln_b, norm_xf_bf, cLT);
  hipLaunchKernelGGL(ln_rows_kernel, dim3(cB*cKR), dim3(TPB), 0, stream, re_motion, rn2_g, rn2_b, norm_rm_bf, cD);
  hipLaunchKernelGGL(ln_refeat_kernel, dim3(cB*cKR), dim3(TPB), 0, stream, re_motion, re_text, rn1_g, rn1_b, norm_rf_bf);

  // 3) projection GEMMs
  gemm(norm_x_bf, WqT, query_f, bq, nullptr,
       cB*cT, cD, cD, cD, cD, cD, 0, 0, 0, 1);
  gemm(norm_xf_bf, WktT, key_f, bkt, nullptr,
       cNT, cD, cLT, cLT, cLT, cD,
       (long long)cNT*cLT, 0, (long long)cNTOK*cD, cB);
  gemm(norm_rf_bf, WkrT, key_f + (size_t)cNT*cD, bkr, nullptr,
       cKR, cD, cD2, cD2, cD2, cD,
       (long long)cKR*cD2, 0, (long long)cNTOK*cD, cB);
  gemm(norm_x_bf, WkmT, key_f + (size_t)(cNT+cKR)*cD, bkm, nullptr,
       cT, cD, cD, cD, cD, cD,
       (long long)cT*cD, 0, (long long)cNTOK*cD, cB);
  gemm(norm_xf_bf, WvtT, value_f, bvt, nullptr,
       cNT, cD, cLT, cLT, cLT, cD,
       (long long)cNT*cLT, 0, (long long)cNTOK*cD, cB);
  gemm(norm_rm_bf, WvrT, value_f + (size_t)cNT*cD, bvr, nullptr,
       cKR, cD, cD, cD, cD, cD,
       (long long)cKR*cD, 0, (long long)cNTOK*cD, cB);
  gemm(norm_x_bf, WvmT, value_f + (size_t)(cNT+cKR)*cD, bvm, nullptr,
       cT, cD, cD, cD, cD, cD,
       (long long)cT*cD, 0, (long long)cNTOK*cD, cB);

  // 4) masks
  {
    size_t tot = (size_t)cB*cNTOK*cD;
    hipLaunchKernelGGL(keymask_kernel, dim3((unsigned)cdiv(tot, TPB)), dim3(TPB), 0, stream, key_f, cond, re_mask, src_mask);
    hipLaunchKernelGGL(valmask_kernel, dim3((unsigned)cdiv(tot, TPB)), dim3(TPB), 0, stream, value_f, cond, re_mask, src_mask);
  }

  // 5) softmaxes + layout transforms (pad K to 2144 with zeros)
  hipLaunchKernelGGL(qsoftmax_kernel, dim3(cB*cT*cH/8), dim3(TPB), 0, stream, query_f, q_bf);
  hipMemsetAsync(kt_bf, 0, (size_t)cB*cH*cDH*cKPAD*2, stream);
  hipMemsetAsync(vt_bf, 0, (size_t)cB*cH*cDH*cKPAD*2, stream);
  hipLaunchKernelGGL(ksoftmax_kernel, dim3(cD/TPB, cB), dim3(TPB), 0, stream, key_f, kt_bf);
  {
    size_t tot = (size_t)cB*cNTOK*cD;
    hipLaunchKernelGGL(vtranspose_kernel, dim3((unsigned)cdiv(tot, TPB)), dim3(TPB), 0, stream, value_f, vt_bf);
  }

  // 6) att[b,h] = k^T (128 x Npad) * v (Npad x 128), batched over 64
  gemm(kt_bf, vt_bf, att_f, nullptr, nullptr,
       cDH, cDH, cKPAD, cKPAD, cKPAD, cDH,
       (long long)cDH*cKPAD, (long long)cDH*cKPAD, (long long)cDH*cDH, cB*cH);
  {
    size_t tot = (size_t)cB*cH*cDH*cDH;
    hipLaunchKernelGGL(atttrans_kernel, dim3((unsigned)cdiv(tot, TPB)), dim3(TPB), 0, stream, att_f, attT_bf);
  }

  // 7) y[b,h] = q (T x 128) * att (128 x 128), batched over 64 -> permute to (B,T,D)
  gemm(q_bf, attT_bf, ytmp_f, nullptr, nullptr,
       cT, cDH, cDH, cDH, cDH, cDH,
       (long long)cT*cDH, (long long)cDH*cDH, (long long)cT*cDH, cB*cH);
  {
    size_t tot = (size_t)cB*cH*cT*cDH;
    hipLaunchKernelGGL(ypermute_kernel, dim3((unsigned)cdiv(tot, TPB)), dim3(TPB), 0, stream, ytmp_f, y_f);
  }

  // 8) stylization
  hipLaunchKernelGGL(siluemb_kernel, dim3((unsigned)cdiv((size_t)cB*cE, TPB)), dim3(TPB), 0, stream, emb, se_f);
  hipLaunchKernelGGL(eo_kernel, dim3(cD2/TPB, cB), dim3(TPB), 0, stream, se_f, We, be, eo_f);
  hipLaunchKernelGGL(ln_film_silu_kernel, dim3(cB*cT), dim3(TPB), 0, stream, y_f, sln_g, sln_b, eo_f, h_bf);

  // 9) out = x + h @ Wo + bo  (residual fused into GEMM epilogue)
  gemm(h_bf, WoT, out, bo, x,
       cB*cT, cD, cD, cD, cD, cD, 0, 0, 0, 1);
}
